// MovieLayer_6846177870359
// MI455X (gfx1250) — matrix operand order repaired and checked by reading, not compiled, not run on hardware
//
#include <hip/hip_runtime.h>

// ---------------------------------------------------------------------------
// MovieLayer: embedding gather (19 nodes x 128 feat) + GEMM (16384x512x2432)
// CDNA5 / gfx1250: bf16 WMMA 16x16x32 (f32 accum), double-buffered LDS,
// software-pipelined: A-gather global loads + async B copies for node nd+1
// are in flight while the 32 WMMAs per wave for node nd execute. bf16 conversion via
// native __bf16 casts (hardware cvt) instead of integer RNE emulation.
// ---------------------------------------------------------------------------

#define N_NODES 19
#define NI      128
#define NO      512
#define VOCAB   20000
#define BS      16384
#define FAN_IN  (N_NODES * NI)   // 2432

#define M_BLK   128
#define N_BLK   256
#define THREADS 512              // 16 waves of 32
#define KPAD    136              // 128 + 8 ushort pad -> 272B row stride (4-bank rotate)

typedef __attribute__((ext_vector_type(16))) __bf16 v16bf;
typedef __attribute__((ext_vector_type(8)))  float  v8f;

union PackBF4 {                  // 4 f32 -> 4 bf16 packed in 8 bytes
  __bf16 h[4];
  uint2  u;
};

__device__ __forceinline__ uint2 cvt4_bf16(float4 v) {
  PackBF4 p;
  p.h[0] = (__bf16)v.x;          // RNE casts -> native v_cvt on gfx1250
  p.h[1] = (__bf16)v.y;
  p.h[2] = (__bf16)v.z;
  p.h[3] = (__bf16)v.w;
  return p.u;
}

// ---- prep: fc_w f32 -> bf16 into workspace --------------------------------
__global__ void wcvt_kernel(const float4* __restrict__ w,
                            uint2* __restrict__ o, int n4) {
  int i = blockIdx.x * blockDim.x + threadIdx.x;
  if (i < n4) o[i] = cvt4_bf16(w[i]);
}

// ---- main: pipelined gather/async-copy + WMMA GEMM ------------------------
__global__ __launch_bounds__(THREADS, 1)
void movie_gemm_kernel(const long long* __restrict__ feature,  // [BS][19] i64
                       const float*     __restrict__ emb,      // [19][20000][128] f32
                       const unsigned short* __restrict__ wbf, // [512][2432] bf16
                       const float*     __restrict__ bias,     // [512] f32
                       float*           __restrict__ out) {    // [BS][512] f32
  // double-buffered tiles: 2*(34KB + 68KB) = 204KB of the 320KB WGP LDS
  __shared__ __align__(16) unsigned short As[2][M_BLK][KPAD];
  __shared__ __align__(16) unsigned short Bs[2][N_BLK][KPAD];

  const int tid  = threadIdx.x;
  const int lane = tid & 31;
  const int wave = tid >> 5;            // 0..15
  const int wm   = wave & 3;            // rows [wm*32, wm*32+32)
  const int wn   = wave >> 2;           // cols [wn*64, wn*64+64)
  const int m0   = blockIdx.y * M_BLK;
  const int n0   = blockIdx.x * N_BLK;

  const int l15    = lane & 15;
  const int hi     = lane >> 4;         // half-wave id
  const int khalfA = hi << 3;           // A frag: K run base 0 / 8

  // ---- per-thread roles for the fill stages ----
  const int arow = tid >> 2;            // A: 0..127, 32 floats each
  const int aq   = tid & 3;
  const int bn   = tid >> 1;            // B: 0..255, 128B each
  const int bh   = tid & 1;

  float4 areg[8];                       // staged gather (in-flight loads)

  auto loadA = [&](int nd) {            // issue global loads only
    long long idx = feature[(long long)(m0 + arow) * N_NODES + nd];
    const float4* src =
        (const float4*)(emb + ((long long)nd * VOCAB + idx) * (long long)NI) + aq * 8;
#pragma unroll
    for (int j = 0; j < 8; ++j) areg[j] = src[j];
  };

  auto storeA = [&](int buf) {          // native cvt to bf16 + ds_store
    uint2* dst = (uint2*)&As[buf][arow][aq * 32];
#pragma unroll
    for (int j = 0; j < 8; ++j) dst[j] = cvt4_bf16(areg[j]);
  };

  auto issueB = [&](int nd, int buf) {  // async global->LDS, ASYNCcnt in flight
    const unsigned gbyte =
        ((unsigned)(n0 + bn) * FAN_IN + (unsigned)(nd * NI + bh * 64)) * 2u;
    // flat LDS pointers carry the byte offset in addr[31:0] (LDS aperture)
    const unsigned lbyte = (unsigned)(unsigned long long)&Bs[buf][bn][bh * 64];
#pragma unroll
    for (int j = 0; j < 8; ++j) {
      asm volatile("global_load_async_to_lds_b128 %0, %1, %2"
                   :: "v"(lbyte + j * 16u), "v"(gbyte + j * 16u), "s"(wbf)
                   : "memory");
    }
  };

  v8f acc[2][4];
  const v8f vzero = {0.f, 0.f, 0.f, 0.f, 0.f, 0.f, 0.f, 0.f};
#pragma unroll
  for (int mi = 0; mi < 2; ++mi)
#pragma unroll
    for (int ni = 0; ni < 4; ++ni) acc[mi][ni] = vzero;

  auto compute = [&](int buf) {
#pragma unroll
    for (int ks = 0; ks < 4; ++ks) {
      union Frag { v16bf v; uint4 q[2]; };
      Frag a[2], b[4];
      // A 16x32 bf16 frag: lanes 0-15 K runs [0..7]&[16..23], lanes 16-31
      // [8..15]&[24..31] (ISA 7.12.2)
      const int kbA = ks * 32 + khalfA;
#pragma unroll
      for (int mi = 0; mi < 2; ++mi) {
        const unsigned short* p = &As[buf][wm * 32 + mi * 16 + l15][kbA];
        a[mi].q[0] = *(const uint4*)p;
        a[mi].q[1] = *(const uint4*)(p + 16);
      }
      // B 32x16 bf16 frag: lane = column, lanes 0-15 K [0..7]&[16..23], 16-31 [8..15]&[24..31]
      const int kbB = ks * 32 + (hi << 3);
#pragma unroll
      for (int ni = 0; ni < 4; ++ni) {
        const unsigned short* p = &Bs[buf][wn * 64 + ni * 16 + l15][kbB];
        b[ni].q[0] = *(const uint4*)p;
        b[ni].q[1] = *(const uint4*)(p + 16);
      }
#pragma unroll
      for (int ni = 0; ni < 4; ++ni)
#pragma unroll
        for (int mi = 0; mi < 2; ++mi)
          acc[mi][ni] = __builtin_amdgcn_wmma_f32_16x16x32_bf16(
              false, a[mi].v, false, b[ni].v, (short)0, acc[mi][ni], false, false);
    }
  };

  // ---- prologue: fill buffer 0 for node 0 ----
  loadA(0);
  issueB(0, 0);
  storeA(0);
  asm volatile("s_wait_asynccnt 0x0" ::: "memory");
  __syncthreads();

  // ---- pipelined main loop over nodes ----
  for (int nd = 0; nd < N_NODES; ++nd) {
    const int  cur = nd & 1;
    const bool pf  = (nd + 1) < N_NODES;
    if (pf) {
      loadA(nd + 1);           // global loads in flight across compute
      issueB(nd + 1, cur ^ 1); // async copies in flight across compute
    }
    compute(cur);              // 32 WMMAs per wave hide the prefetch latency
    if (pf) {
      storeA(cur ^ 1);         // waits LOADcnt here, cvt + ds_store
      asm volatile("s_wait_asynccnt 0x0" ::: "memory");
    }
    __syncthreads();           // publish buffer cur^1, retire readers of cur
  }

  // ---- epilogue: C layout = lane(l): col=l&15, rows v + 8*(l>=16) ----
#pragma unroll
  for (int ni = 0; ni < 4; ++ni) {
    const int col = n0 + wn * 64 + ni * 16 + l15;
    const float bv = bias[col];
#pragma unroll
    for (int mi = 0; mi < 2; ++mi) {
      const int rbase = m0 + wm * 32 + mi * 16 + (hi << 3);
#pragma unroll
      for (int v = 0; v < 8; ++v) {
        out[(long long)(rbase + v) * NO + col] = acc[mi][ni][v] + bv;
      }
    }
  }
}

extern "C" void kernel_launch(void* const* d_in, const int* in_sizes, int n_in,
                              void* d_out, int out_size, void* d_ws, size_t ws_size,
                              hipStream_t stream) {
  const long long* feature = (const long long*)d_in[0];   // int64
  const float*     emb     = (const float*)d_in[1];
  const float*     fc_w    = (const float*)d_in[2];
  const float*     fc_b    = (const float*)d_in[3];
  float*           out     = (float*)d_out;
  unsigned short*  wbf     = (unsigned short*)d_ws;       // bf16 weights, 2.49 MB

  const int n4 = (NO * FAN_IN) / 4;                       // 311296 float4s
  wcvt_kernel<<<(n4 + 255) / 256, 256, 0, stream>>>((const float4*)fc_w,
                                                    (uint2*)wbf, n4);

  dim3 grid(NO / N_BLK, BS / M_BLK);                      // (2, 128)
  movie_gemm_kernel<<<grid, THREADS, 0, stream>>>(feature, emb, wbf, fc_b, out);
}
